// TextClassifier_33277406610125
// MI455X (gfx1250) — matrix operand order repaired and checked by reading, not compiled, not run on hardware
//
#include <hip/hip_runtime.h>
#include <hip/hip_bf16.h>
#include <math.h>

typedef __attribute__((ext_vector_type(16))) _Float16 v16h;
typedef __attribute__((ext_vector_type(8)))  float    v8f;

#define EMBED     512
#define HEADS     8
#define HEAD_DIM  64
#define WINDOW    128
#define NUM_CLASS 20
#define BB        4
#define SS        2048
#define QKV_N     (3 * EMBED)
#define BAND_MAX  288   // 272-wide band padded to 18 chunks of 16

// WMMA fragments: per lane, A(16x32 f16) and B(32x16 f16) = two contiguous
// 8-f16 runs (k = 8*half.., 16+8*half..) -> load as 2x uint4 (b128).
union FragAB { v16h h; unsigned u[8]; uint4 q[2]; };
union FragC  { v8f  f; float    x[8]; };
union HalfPair { unsigned u; _Float16 h[2]; };

// A-matrix packed-dword K position (scalar fallback / LDS builds)
__device__ __forceinline__ int a_kpos(int e, int half) {
    return (e < 4 ? 2 * e : 16 + 2 * (e - 4)) + 8 * half;
}
// B-matrix packed-dword K position
__device__ __forceinline__ int b_kpos(int v, int half) { return (v < 4 ? 2 * v : 16 + 2 * (v - 4)) + 8 * half; }

// ---------------------------------------------------------------------------
// Elementwise helpers
// ---------------------------------------------------------------------------
__global__ void f32_to_f16_kernel(const float* __restrict__ src,
                                  _Float16* __restrict__ dst, int n) {
    int i = blockIdx.x * blockDim.x + threadIdx.x;
    if (i < n) dst[i] = (_Float16)src[i];
}

__global__ void embed_gather_f16(const int* __restrict__ text,
                                 const float* __restrict__ emb,
                                 _Float16* __restrict__ x, int total) {
    int i = blockIdx.x * blockDim.x + threadIdx.x;
    if (i >= total) return;
    int row = i / EMBED, col = i - row * EMBED;
    int tok = text[row];
    x[i] = (_Float16)emb[(size_t)tok * EMBED + col];
}

// ---------------------------------------------------------------------------
// f16 WMMA GEMM, 4x2 register-blocked: each wave owns a 64x32 output tile.
// C[M,N] = A[M,K] * W[N,K]^T + bias
// ---------------------------------------------------------------------------
#define TM 4
#define TN 2

template <bool OUT_F32>
__global__ __launch_bounds__(32)
void gemm_f16_wmma(const _Float16* __restrict__ A,   // M x K row-major
                   const _Float16* __restrict__ W,   // N x K row-major
                   const float* __restrict__ bias,   // N
                   void* __restrict__ Cout,          // M x N
                   int M, int N, int K) {
    const int lane = threadIdx.x & 31;
    const int half = lane >> 4;
    const int mn   = lane & 15;
    const int n0   = blockIdx.x * (16 * TN);
    const int m0   = blockIdx.y * (16 * TM);

    FragC c[TM][TN];
#pragma unroll
    for (int tm = 0; tm < TM; ++tm)
#pragma unroll
        for (int tn = 0; tn < TN; ++tn)
#pragma unroll
            for (int r = 0; r < 8; ++r) c[tm][tn].x[r] = 0.f;

    const _Float16* arow[TM];
    const _Float16* wrow[TN];
#pragma unroll
    for (int tm = 0; tm < TM; ++tm)
        arow[tm] = A + (size_t)(m0 + tm * 16 + mn) * K;
#pragma unroll
    for (int tn = 0; tn < TN; ++tn)
        wrow[tn] = W + (size_t)(n0 + tn * 16 + mn) * K;

    for (int k0 = 0; k0 < K; k0 += 32) {
        FragAB a[TM], b[TN];
#pragma unroll
        for (int tm = 0; tm < TM; ++tm) {
            a[tm].q[0] = *(const uint4*)(arow[tm] + k0 + 8 * half);
            a[tm].q[1] = *(const uint4*)(arow[tm] + k0 + 16 + 8 * half);
        }
#pragma unroll
        for (int tn = 0; tn < TN; ++tn) {
            b[tn].q[0] = *(const uint4*)(wrow[tn] + k0 + 8 * half);
            b[tn].q[1] = *(const uint4*)(wrow[tn] + k0 + 16 + 8 * half);
        }
        // prefetch next K-step (lowers to global_prefetch_b8)
        __builtin_prefetch(arow[0] + k0 + 64, 0, 0);
        __builtin_prefetch(wrow[0] + k0 + 64, 0, 0);
#pragma unroll
        for (int tm = 0; tm < TM; ++tm)
#pragma unroll
            for (int tn = 0; tn < TN; ++tn)
                c[tm][tn].f = __builtin_amdgcn_wmma_f32_16x16x32_f16(
                    false, a[tm].h, false, b[tn].h, (short)0, c[tm][tn].f,
                    false, false);
    }

#pragma unroll
    for (int tn = 0; tn < TN; ++tn) {
        const int n  = n0 + tn * 16 + mn;
        const float bv = bias[n];
#pragma unroll
        for (int tm = 0; tm < TM; ++tm)
#pragma unroll
            for (int r = 0; r < 8; ++r) {
                const int m = m0 + tm * 16 + r + 8 * half;
                const float val = c[tm][tn].x[r] + bv;
                if (OUT_F32) ((float*)Cout)[(size_t)m * N + n] = val;
                else         ((_Float16*)Cout)[(size_t)m * N + n] = (_Float16)val;
            }
    }
}

// ---------------------------------------------------------------------------
// Banded attention: one wave per (b, h, 16-query block).
// qkv layout: (B*S) x 1536, cols [0,512)=Q [512,1024)=K [1024,1536)=V
// ---------------------------------------------------------------------------
__global__ __launch_bounds__(32)
void banded_attention(const _Float16* __restrict__ qkv,
                      _Float16* __restrict__ attn) {  // (B*S) x 512
    __shared__ float sc[16][BAND_MAX];

    const int lane = threadIdx.x & 31;
    const int half = lane >> 4;
    const int mn   = lane & 15;
    const int i0   = blockIdx.x * 16;
    const int h    = blockIdx.y;
    const int b    = blockIdx.z;

    const int jlo = max(0, i0 - WINDOW);
    const int jhi = min(SS, i0 + 16 + WINDOW);
    const int nch = (jhi - jlo + 15) >> 4;

    // init whole band buffer to -inf so pad columns softmax to exactly 0
    for (int t = threadIdx.x; t < 16 * BAND_MAX; t += 32)
        ((float*)sc)[t] = -INFINITY;
    __syncthreads();

    // Q fragments (A-matrix 16x32, two halves of d=64), b128 loads
    FragAB qa[2];
    {
        const _Float16* qrow =
            qkv + (size_t)(b * SS + i0 + mn) * QKV_N + h * HEAD_DIM;
        qa[0].q[0] = *(const uint4*)(qrow + 8 * half);
        qa[0].q[1] = *(const uint4*)(qrow + 16 + 8 * half);
        qa[1].q[0] = *(const uint4*)(qrow + 32 + 8 * half);
        qa[1].q[1] = *(const uint4*)(qrow + 48 + 8 * half);
    }

    // scores: 16x16 tile per 16-key chunk, 2 WMMAs (K = d = 64)
    for (int ch = 0; ch < nch; ++ch) {
        const int j0 = jlo + ch * 16;
        const int jr = min(j0 + mn, SS - 1);  // clamp; masked below
        const _Float16* krow =
            qkv + (size_t)(b * SS + jr) * QKV_N + EMBED + h * HEAD_DIM;
        FragAB kb0, kb1;
        kb0.q[0] = *(const uint4*)(krow + 8 * half);
        kb0.q[1] = *(const uint4*)(krow + 16 + 8 * half);
        kb1.q[0] = *(const uint4*)(krow + 32 + 8 * half);
        kb1.q[1] = *(const uint4*)(krow + 48 + 8 * half);

        FragC s;
#pragma unroll
        for (int r = 0; r < 8; ++r) s.x[r] = 0.f;
        s.f = __builtin_amdgcn_wmma_f32_16x16x32_f16(false, qa[0].h, false, kb0.h,
                                                     (short)0, s.f, false, false);
        s.f = __builtin_amdgcn_wmma_f32_16x16x32_f16(false, qa[1].h, false, kb1.h,
                                                     (short)0, s.f, false, false);
        const int j = j0 + mn;
#pragma unroll
        for (int r = 0; r < 8; ++r) {
            const int i = i0 + r + 8 * half;
            const bool ok = (j < jhi) && (j >= i - WINDOW) && (j <= i + WINDOW);
            sc[r + 8 * half][ch * 16 + mn] = ok ? s.x[r] * 0.125f : -INFINITY;
        }
    }
    __syncthreads();

    // softmax: lanes 0..15 each own one row of the band
    if (lane < 16) {
        float mx = -INFINITY;
        for (int t = 0; t < BAND_MAX; ++t) mx = fmaxf(mx, sc[lane][t]);
        float sum = 0.f;
        for (int t = 0; t < BAND_MAX; ++t) {
            const float p = expf(sc[lane][t] - mx);  // exp(-inf - mx) == 0
            sc[lane][t] = p;
            sum += p;
        }
        const float inv = 1.f / sum;
        for (int t = 0; t < BAND_MAX; ++t) sc[lane][t] *= inv;
    }
    __syncthreads();

    // P @ V : 32-key pairs x 4 d-groups of 16
    FragC acc[4];
#pragma unroll
    for (int g = 0; g < 4; ++g)
#pragma unroll
        for (int r = 0; r < 8; ++r) acc[g].x[r] = 0.f;

    const _Float16* vbase = qkv + 2 * EMBED + h * HEAD_DIM;
    for (int p = 0; p < BAND_MAX / 32; ++p) {
        FragAB pa;  // prob fragment (A-matrix), converted f32 -> f16
#pragma unroll
        for (int e = 0; e < 8; ++e) {
            const int kp = p * 32 + a_kpos(e, half);
            HalfPair hp;
            hp.h[0] = (_Float16)sc[mn][kp];
            hp.h[1] = (_Float16)sc[mn][kp + 1];
            pa.u[e] = hp.u;
        }
#pragma unroll
        for (int g = 0; g < 4; ++g) {
            FragAB vb;
#pragma unroll
            for (int e = 0; e < 8; ++e) {
                const int k  = p * 32 + b_kpos(e, half);
                const int ja = min(jlo + k, SS - 1);      // pad keys: prob==0
                const int jb = min(jlo + k + 1, SS - 1);
                HalfPair hp;
                hp.h[0] = vbase[(size_t)(b * SS + ja) * QKV_N + g * 16 + mn];
                hp.h[1] = vbase[(size_t)(b * SS + jb) * QKV_N + g * 16 + mn];
                vb.u[e] = hp.u;
            }
            acc[g].f = __builtin_amdgcn_wmma_f32_16x16x32_f16(
                false, pa.h, false, vb.h, (short)0, acc[g].f, false, false);
        }
    }

#pragma unroll
    for (int g = 0; g < 4; ++g)
#pragma unroll
        for (int r = 0; r < 8; ++r) {
            const int i = i0 + r + 8 * half;
            const int d = g * 16 + mn;
            attn[(size_t)(b * SS + i) * EMBED + h * HEAD_DIM + d] =
                (_Float16)acc[g].x[r];
        }
}

// ---------------------------------------------------------------------------
// Max-pool over sequence, scalar MLP
// ---------------------------------------------------------------------------
__global__ void maxpool_seq(const float* __restrict__ proj,
                            float* __restrict__ ctx) {
    int idx = blockIdx.x * blockDim.x + threadIdx.x;
    if (idx >= BB * EMBED) return;
    const int b = idx / EMBED, e = idx - b * EMBED;
    float m = -INFINITY;
    for (int s = 0; s < SS; ++s)
        m = fmaxf(m, proj[(size_t)(b * SS + s) * EMBED + e]);
    ctx[idx] = m;
}

__global__ void dense_f32(const float* __restrict__ X, const float* __restrict__ W,
                          const float* __restrict__ bias, float* __restrict__ Y,
                          int Bn, int N, int K, int leaky) {
    int idx = blockIdx.x * blockDim.x + threadIdx.x;
    if (idx >= Bn * N) return;
    const int bi = idx / N, n = idx - bi * N;
    float s = bias[n];
    for (int k = 0; k < K; ++k) s += X[bi * K + k] * W[n * K + k];
    if (leaky) s = (s > 0.f) ? s : 0.01f * s;
    Y[idx] = s;
}

// ---------------------------------------------------------------------------
extern "C" void kernel_launch(void* const* d_in, const int* in_sizes, int n_in,
                              void* d_out, int out_size, void* d_ws, size_t ws_size,
                              hipStream_t stream) {
    const int*   text  = (const int*)d_in[0];
    const float* emb   = (const float*)d_in[1];
    const float* in_w  = (const float*)d_in[2];
    const float* in_b  = (const float*)d_in[3];
    const float* out_w = (const float*)d_in[4];
    const float* out_b = (const float*)d_in[5];
    const float* w1 = (const float*)d_in[6];  const float* b1 = (const float*)d_in[7];
    const float* w2 = (const float*)d_in[8];  const float* b2 = (const float*)d_in[9];
    const float* w3 = (const float*)d_in[10]; const float* b3 = (const float*)d_in[11];
    const float* w4 = (const float*)d_in[12]; const float* b4 = (const float*)d_in[13];

    char* ws = (char*)d_ws;
    size_t off = 0;
    auto alloc = [&](size_t bytes) {
        void* p = ws + off;
        off = (off + bytes + 255) & ~(size_t)255;
        return p;
    };
    const int MROWS = BB * SS;  // 8192
    _Float16* xf16   = (_Float16*)alloc((size_t)MROWS * EMBED * 2);
    _Float16* inw16  = (_Float16*)alloc((size_t)QKV_N * EMBED * 2);
    _Float16* outw16 = (_Float16*)alloc((size_t)EMBED * EMBED * 2);
    _Float16* qkv16  = (_Float16*)alloc((size_t)MROWS * QKV_N * 2);
    _Float16* attn16 = (_Float16*)alloc((size_t)MROWS * EMBED * 2);
    float*    projf  = (float*)alloc((size_t)MROWS * EMBED * 4);
    float*    ctxf   = (float*)alloc((size_t)BB * EMBED * 4);
    float*    h1f    = (float*)alloc((size_t)BB * 512 * 4);
    float*    h2f    = (float*)alloc((size_t)BB * 256 * 4);
    float*    h3f    = (float*)alloc((size_t)BB * 128 * 4);

    // weight + activation conversion to f16
    {
        int n = QKV_N * EMBED;
        f32_to_f16_kernel<<<(n + 255) / 256, 256, 0, stream>>>(in_w, inw16, n);
        n = EMBED * EMBED;
        f32_to_f16_kernel<<<(n + 255) / 256, 256, 0, stream>>>(out_w, outw16, n);
        n = MROWS * EMBED;
        embed_gather_f16<<<(n + 255) / 256, 256, 0, stream>>>(text, emb, xf16, n);
    }

    // QKV projection: (8192 x 512) @ (512 x 1536) -> f16
    {
        dim3 grid(QKV_N / (16 * TN), MROWS / (16 * TM));
        gemm_f16_wmma<false><<<grid, 32, 0, stream>>>(xf16, inw16, in_b, qkv16,
                                                      MROWS, QKV_N, EMBED);
    }

    // banded attention
    {
        dim3 grid(SS / 16, HEADS, BB);
        banded_attention<<<grid, 32, 0, stream>>>(qkv16, attn16);
    }

    // output projection -> f32
    {
        dim3 grid(EMBED / (16 * TN), MROWS / (16 * TM));
        gemm_f16_wmma<true><<<grid, 32, 0, stream>>>(attn16, outw16, out_b, projf,
                                                     MROWS, EMBED, EMBED);
    }

    // max-pool over sequence
    maxpool_seq<<<(BB * EMBED + 255) / 256, 256, 0, stream>>>(projf, ctxf);

    // MLP head (tiny, scalar f32)
    dense_f32<<<(BB * 512 + 255) / 256, 256, 0, stream>>>(ctxf, w1, b1, h1f, BB, 512, 512, 1);
    dense_f32<<<(BB * 256 + 255) / 256, 256, 0, stream>>>(h1f, w2, b2, h2f, BB, 256, 512, 1);
    dense_f32<<<(BB * 128 + 255) / 256, 256, 0, stream>>>(h2f, w3, b3, h3f, BB, 128, 256, 1);
    dense_f32<<<1, 256, 0, stream>>>(h3f, w4, b4, (float*)d_out, BB, NUM_CLASS, 128, 0);
}
